// MyClassifier_19662360281115
// MI455X (gfx1250) — compile-verified, matrix operand order checked by reading, not run on hardware
//
#include <hip/hip_runtime.h>

// ---------------------------------------------------------------------------
// Shapes (hardcoded per reference)
// ---------------------------------------------------------------------------
#define B_  32
#define S_  2048
#define T_  64
#define C_  512
#define L_  16
#define D_  768
#define H_  8
#define DH_ 96
#define BT_ (B_ * T_)   // 2048
#define BC_ (B_ * C_)   // 16384
#define TPC_ (T_ + C_)  // 576

typedef __attribute__((ext_vector_type(16))) __bf16 v16bf;
typedef __attribute__((ext_vector_type(8)))  __bf16 v8bf;
typedef __attribute__((ext_vector_type(8)))  float  v8f;

// ---------------------------------------------------------------------------
// Load a 16-element bf16 WMMA fragment half-row: elements 0..7 contiguous at
// base, elements 8..15 contiguous at base+16 (per CDNA5 16-bit A/B layout).
// ---------------------------------------------------------------------------
__device__ __forceinline__ v16bf load_frag16(const __bf16* base) {
  v8bf lo = *(const v8bf*)(base);
  v8bf hi = *(const v8bf*)(base + 16);
  return __builtin_shufflevector(lo, hi, 0, 1, 2, 3, 4, 5, 6, 7,
                                         8, 9, 10, 11, 12, 13, 14, 15);
}

__device__ __forceinline__ v8f wmma_bf16(v16bf a, v16bf b, v8f c) {
  return __builtin_amdgcn_wmma_f32_16x16x32_bf16(
      false, a, false, b, (short)0, c, false, false);
}

// ---------------------------------------------------------------------------
// Ragged gather + pool: out[r, :] = sum_l seq[b, ids[r*L+l], :]  (bf16 out)
// one block (256 threads) per pooled row
// ---------------------------------------------------------------------------
__global__ void gather_pool_kernel(const float* __restrict__ seq,
                                   const int* __restrict__ ids,
                                   __bf16* __restrict__ outb, int perB) {
  int r = blockIdx.x;
  int b = r / perB;
  const int* idp = ids + (size_t)r * L_;
  int lid[L_];
#pragma unroll
  for (int l = 0; l < L_; ++l) lid[l] = idp[l];
  for (int d = threadIdx.x; d < D_; d += blockDim.x) {
    float s = 0.f;
#pragma unroll
    for (int l = 0; l < L_; ++l)
      s += seq[((size_t)b * S_ + lid[l]) * D_ + d];
    outb[(size_t)r * D_ + d] = (__bf16)s;
  }
}

// ---------------------------------------------------------------------------
// Weight conversion: out[n*K + k] = W[k*N + n] (transpose) or W[n*K + k]
// produces the [N, K] bf16 layout so B fragments load contiguously in K
// ---------------------------------------------------------------------------
__global__ void conv_weight_kernel(const float* __restrict__ W,
                                   __bf16* __restrict__ out,
                                   int K, int N, int transpose) {
  int i = blockIdx.x * blockDim.x + threadIdx.x;
  if (i >= N * K) return;
  int n = i / K, k = i % K;
  float v = transpose ? W[(size_t)k * N + n] : W[(size_t)n * K + k];
  out[i] = (__bf16)v;
}

__global__ void f32_to_bf16_kernel(const float* __restrict__ in,
                                   __bf16* __restrict__ out, int n) {
  int i = blockIdx.x * blockDim.x + threadIdx.x;
  if (i < n) out[i] = (__bf16)in[i];
}

// ---------------------------------------------------------------------------
// WMMA bf16 GEMM: C[M,N] = act(A[M,K] @ Bw^T + bias), Bw stored [N,K].
// 256 threads = 8 waves; each wave owns a 32x32 output tile: 2 A fragments x
// 2 B fragments -> 4 v_wmma_f32_16x16x32_bf16 per K-step, so each loaded
// fragment is used twice (2 b128 loads per WMMA instead of 4). Early exit is
// wave-uniform so EXEC is all-ones for every WMMA.
// ---------------------------------------------------------------------------
__device__ __forceinline__ void store_subtile(float* __restrict__ Cf,
                                              __bf16* __restrict__ Cb,
                                              v8f acc, int rowBase, int col,
                                              int N, float bs, int act) {
#pragma unroll
  for (int r = 0; r < 8; ++r) {
    float v = acc[r] + bs;
    if (act) v = (v >= 0.f) ? v : 0.01f * v;
    size_t o = (size_t)(rowBase + r) * N + col;
    if (Cf) Cf[o] = v;
    if (Cb) Cb[o] = (__bf16)v;
  }
}

__global__ void gemm_bf16_wmma_kernel(const __bf16* __restrict__ A,
                                      const __bf16* __restrict__ Bw,
                                      const float* __restrict__ bias,
                                      float* __restrict__ Cf,
                                      __bf16* __restrict__ Cb,
                                      int M, int N, int K, int act) {
  int lane = threadIdx.x & 31;
  int wave = threadIdx.x >> 5;
  int tiles_n = N >> 5;                    // 32-wide tiles
  int tile = blockIdx.x * 8 + wave;
  if (tile >= (M >> 5) * tiles_n) return;  // uniform per wave
  int mt = tile / tiles_n;
  int nt = tile % tiles_n;
  int laneLo = lane & 15;
  int laneHi = lane >> 4;

  const __bf16* aPtr0 = A  + (size_t)(mt * 32 + laneLo) * K + laneHi * 8;
  const __bf16* aPtr1 = aPtr0 + (size_t)16 * K;
  const __bf16* bPtr0 = Bw + (size_t)(nt * 32 + laneLo) * K + laneHi * 8;
  const __bf16* bPtr1 = bPtr0 + (size_t)16 * K;

  v8f acc00 = {}, acc01 = {}, acc10 = {}, acc11 = {};
  for (int k0 = 0; k0 < K; k0 += 32) {
    v16bf a0 = load_frag16(aPtr0 + k0);
    v16bf a1 = load_frag16(aPtr1 + k0);
    v16bf b0 = load_frag16(bPtr0 + k0);
    v16bf b1 = load_frag16(bPtr1 + k0);
    acc00 = wmma_bf16(a0, b0, acc00);
    acc01 = wmma_bf16(a0, b1, acc01);
    acc10 = wmma_bf16(a1, b0, acc10);
    acc11 = wmma_bf16(a1, b1, acc11);
  }

  int col0 = nt * 32 + laneLo;
  int col1 = col0 + 16;
  float bs0 = bias ? bias[col0] : 0.f;
  float bs1 = bias ? bias[col1] : 0.f;
  int row0 = mt * 32 + laneHi * 8;
  store_subtile(Cf, Cb, acc00, row0,      col0, N, bs0, act);
  store_subtile(Cf, Cb, acc01, row0,      col1, N, bs1, act);
  store_subtile(Cf, Cb, acc10, row0 + 16, col0, N, bs0, act);
  store_subtile(Cf, Cb, acc11, row0 + 16, col1, N, bs1, act);
}

// ---------------------------------------------------------------------------
// Fused masked cross-attention: one block per (b,t); wave w = head h.
// scores over C=512 cols, mask = (table_of_column[b,c] == t), softmax, ctx.
// ---------------------------------------------------------------------------
__global__ void attention_kernel(const float* __restrict__ q,
                                 const float* __restrict__ k,
                                 const float* __restrict__ v,
                                 const int* __restrict__ toc,
                                 float* __restrict__ ctx) {
  __shared__ float attnS[H_][C_];
  __shared__ float qs[H_][DH_];
  int bt = blockIdx.x;
  int b = bt / T_, t = bt % T_;
  int h = threadIdx.x >> 5, lane = threadIdx.x & 31;

  const float* qrow = q + (size_t)bt * D_ + h * DH_;
#pragma unroll
  for (int j = 0; j < 3; ++j) qs[h][lane * 3 + j] = qrow[lane * 3 + j];
  // qs[h] is produced and consumed by the same wave: LDS ops are in-order.

  const int* tocb = toc + b * C_;
  const float scale = 0.10206207261596575f;  // 1/sqrt(96)
  float sloc[16];
  float smax = -3.4e38f;
#pragma unroll 1
  for (int i = 0; i < 16; ++i) {
    int c = lane + i * 32;
    const float* krow = k + ((size_t)b * C_ + c) * D_ + h * DH_;
    float s = 0.f;
    for (int d = 0; d < DH_; ++d) s += qs[h][d] * krow[d];
    s *= scale;
    if (tocb[c] != t) s = -1e9f;
    sloc[i] = s;
    smax = fmaxf(smax, s);
  }
  for (int off = 16; off; off >>= 1)
    smax = fmaxf(smax, __shfl_xor(smax, off, 32));

  float ssum = 0.f;
#pragma unroll
  for (int i = 0; i < 16; ++i) {
    float e = __expf(sloc[i] - smax);
    attnS[h][lane + i * 32] = e;
    ssum += e;
  }
  for (int off = 16; off; off >>= 1) ssum += __shfl_xor(ssum, off, 32);
  float inv = 1.f / ssum;
#pragma unroll
  for (int i = 0; i < 16; ++i) attnS[h][lane + i * 32] *= inv;

#pragma unroll
  for (int j = 0; j < 3; ++j) {
    int d = lane * 3 + j;
    float acc = 0.f;
    for (int c = 0; c < C_; ++c)
      acc += attnS[h][c] * v[((size_t)b * C_ + c) * D_ + h * DH_ + d];
    ctx[(size_t)bt * D_ + h * DH_ + d] = acc;
  }
}

// ---------------------------------------------------------------------------
// Residual add + L2 row normalize; one block per row
// ---------------------------------------------------------------------------
__global__ void resnorm_kernel(const float* __restrict__ tab,
                               const float* __restrict__ proj,
                               float* __restrict__ outf,
                               __bf16* __restrict__ outb) {
  __shared__ float red[8];
  int r = blockIdx.x;
  int tid = threadIdx.x;
  float vals[3];
  float ssq = 0.f;
#pragma unroll
  for (int j = 0; j < 3; ++j) {
    int d = tid + j * 256;
    float val = tab[(size_t)r * D_ + d] + proj[(size_t)r * D_ + d];
    vals[j] = val;
    ssq += val * val;
  }
  for (int off = 16; off; off >>= 1) ssq += __shfl_xor(ssq, off, 32);
  if ((tid & 31) == 0) red[tid >> 5] = ssq;
  __syncthreads();
  float tot = red[0] + red[1] + red[2] + red[3] +
              red[4] + red[5] + red[6] + red[7];
  float inv = 1.f / fmaxf(sqrtf(tot), 1e-12f);
#pragma unroll
  for (int j = 0; j < 3; ++j) {
    int d = tid + j * 256;
    float val = vals[j] * inv;
    outf[(size_t)r * D_ + d] = val;
    outb[(size_t)r * D_ + d] = (__bf16)val;
  }
}

// ---------------------------------------------------------------------------
// Second head layer: [rows,256] @ [256,2] + b2 -> scattered into out[B,576,2]
// ---------------------------------------------------------------------------
__global__ void head2_kernel(const float* __restrict__ hbuf,
                             const float* __restrict__ W2,
                             const float* __restrict__ b2,
                             float* __restrict__ out,
                             int rows, int perB, int colOffset) {
  int r = blockIdx.x * blockDim.x + threadIdx.x;
  if (r >= rows) return;
  int b = r / perB, i = r % perB;
  float a0 = b2[0], a1 = b2[1];
  const float* hp = hbuf + (size_t)r * 256;
  for (int kk = 0; kk < 256; ++kk) {
    float hv = hp[kk];
    a0 += hv * W2[kk * 2 + 0];
    a1 += hv * W2[kk * 2 + 1];
  }
  size_t o = ((size_t)b * TPC_ + colOffset + i) * 2;
  out[o] = a0;
  out[o + 1] = a1;
}

// ---------------------------------------------------------------------------
// Launcher
// ---------------------------------------------------------------------------
static inline size_t alignUp(size_t x) { return (x + 255) & ~(size_t)255; }

extern "C" void kernel_launch(void* const* d_in, const int* in_sizes, int n_in,
                              void* d_out, int out_size, void* d_ws, size_t ws_size,
                              hipStream_t stream) {
  (void)in_sizes; (void)n_in; (void)out_size; (void)ws_size;
  const float* seq       = (const float*)d_in[0];
  const int*   tab_ids   = (const int*)d_in[1];
  const int*   col_ids   = (const int*)d_in[2];
  const int*   toc       = (const int*)d_in[3];
  const float* W_tab     = (const float*)d_in[4];
  const float* b_tab     = (const float*)d_in[5];
  const float* W_col     = (const float*)d_in[6];
  const float* b_col     = (const float*)d_in[7];
  const float* in_proj_w = (const float*)d_in[8];
  const float* in_proj_b = (const float*)d_in[9];
  const float* out_w     = (const float*)d_in[10];
  const float* out_b     = (const float*)d_in[11];
  const float* W_t1      = (const float*)d_in[12];
  const float* b_t1      = (const float*)d_in[13];
  const float* W_t2      = (const float*)d_in[14];
  const float* b_t2      = (const float*)d_in[15];
  const float* W_c1      = (const float*)d_in[16];
  const float* b_c1      = (const float*)d_in[17];
  const float* W_c2      = (const float*)d_in[18];
  const float* b_c2      = (const float*)d_in[19];
  float* out = (float*)d_out;

  // ---- workspace carve-up ----
  char* p = (char*)d_ws;
  size_t off = 0;
  auto takeB = [&](size_t elems) { __bf16* r = (__bf16*)(p + off); off = alignUp(off + elems * sizeof(__bf16)); return r; };
  auto takeF = [&](size_t elems) { float*  r = (float*)(p + off);  off = alignUp(off + elems * sizeof(float));  return r; };

  __bf16* tabsum_b = takeB((size_t)BT_ * D_);
  __bf16* colsum_b = takeB((size_t)BC_ * D_);
  __bf16* w_tab_b  = takeB((size_t)D_ * D_);
  __bf16* w_col_b  = takeB((size_t)D_ * D_);
  __bf16* wq_b     = takeB((size_t)D_ * D_);
  __bf16* wk_b     = takeB((size_t)D_ * D_);
  __bf16* wv_b     = takeB((size_t)D_ * D_);
  __bf16* wo_b     = takeB((size_t)D_ * D_);
  __bf16* wt1_b    = takeB((size_t)256 * D_);
  __bf16* wc1_b    = takeB((size_t)256 * D_);
  float*  tab_f    = takeF((size_t)BT_ * D_);
  __bf16* tab_b    = takeB((size_t)BT_ * D_);
  __bf16* col_b    = takeB((size_t)BC_ * D_);
  float*  q_f      = takeF((size_t)BT_ * D_);
  float*  k_f      = takeF((size_t)BC_ * D_);
  float*  v_f      = takeF((size_t)BC_ * D_);
  float*  ctx_f    = takeF((size_t)BT_ * D_);
  __bf16* ctx_b    = takeB((size_t)BT_ * D_);
  float*  proj_f   = takeF((size_t)BT_ * D_);
  float*  tabn_f   = takeF((size_t)BT_ * D_);
  __bf16* tabn_b   = takeB((size_t)BT_ * D_);
  float*  ht_f     = takeF((size_t)BT_ * 256);
  float*  hc_f     = takeF((size_t)BC_ * 256);

  const int THREADS = 256;
  auto cdiv = [](int a, int b) { return (a + b - 1) / b; };

  // ---- weight conversions (bf16, [N,K] layout) ----
  int nDD = D_ * D_, nDH = 256 * D_;
  conv_weight_kernel<<<cdiv(nDD, THREADS), THREADS, 0, stream>>>(W_tab, w_tab_b, D_, D_, 1);
  conv_weight_kernel<<<cdiv(nDD, THREADS), THREADS, 0, stream>>>(W_col, w_col_b, D_, D_, 1);
  conv_weight_kernel<<<cdiv(nDD, THREADS), THREADS, 0, stream>>>(in_proj_w,              wq_b, D_, D_, 0);
  conv_weight_kernel<<<cdiv(nDD, THREADS), THREADS, 0, stream>>>(in_proj_w + nDD,        wk_b, D_, D_, 0);
  conv_weight_kernel<<<cdiv(nDD, THREADS), THREADS, 0, stream>>>(in_proj_w + 2 * nDD,    wv_b, D_, D_, 0);
  conv_weight_kernel<<<cdiv(nDD, THREADS), THREADS, 0, stream>>>(out_w, wo_b, D_, D_, 0);
  conv_weight_kernel<<<cdiv(nDH, THREADS), THREADS, 0, stream>>>(W_t1, wt1_b, D_, 256, 1);
  conv_weight_kernel<<<cdiv(nDH, THREADS), THREADS, 0, stream>>>(W_c1, wc1_b, D_, 256, 1);

  // ---- gather + pool ----
  gather_pool_kernel<<<BT_, THREADS, 0, stream>>>(seq, tab_ids, tabsum_b, T_);
  gather_pool_kernel<<<BC_, THREADS, 0, stream>>>(seq, col_ids, colsum_b, C_);

  // ---- WMMA GEMM helper: 32x32 tiles per wave, 8 waves per block ----
  auto gemm = [&](const __bf16* A, const __bf16* Bw, const float* bias,
                  float* Cf, __bf16* Cb, int M, int N, int K, int act) {
    int tiles = (M >> 5) * (N >> 5);
    gemm_bf16_wmma_kernel<<<cdiv(tiles, 8), THREADS, 0, stream>>>(
        A, Bw, bias, Cf, Cb, M, N, K, act);
  };

  // ---- pooled projections (leaky) ----
  gemm(tabsum_b, w_tab_b, b_tab, tab_f, tab_b, BT_, D_, D_, 1);
  gemm(colsum_b, w_col_b, b_col, nullptr, col_b, BC_, D_, D_, 1);

  // ---- q/k/v projections ----
  gemm(tab_b, wq_b, in_proj_b,           q_f, nullptr, BT_, D_, D_, 0);
  gemm(col_b, wk_b, in_proj_b + D_,      k_f, nullptr, BC_, D_, D_, 0);
  gemm(col_b, wv_b, in_proj_b + 2 * D_,  v_f, nullptr, BC_, D_, D_, 0);

  // ---- masked cross-attention ----
  attention_kernel<<<BT_, THREADS, 0, stream>>>(q_f, k_f, v_f, toc, ctx_f);
  f32_to_bf16_kernel<<<cdiv(BT_ * D_, THREADS), THREADS, 0, stream>>>(ctx_f, ctx_b, BT_ * D_);

  // ---- output projection + residual + L2 norm ----
  gemm(ctx_b, wo_b, out_b, proj_f, nullptr, BT_, D_, D_, 0);
  resnorm_kernel<<<BT_, THREADS, 0, stream>>>(tab_f, proj_f, tabn_f, tabn_b);

  // ---- classification heads ----
  gemm(tabn_b, wt1_b, b_t1, ht_f, nullptr, BT_, 256, D_, 1);
  gemm(col_b,  wc1_b, b_c1, hc_f, nullptr, BC_, 256, D_, 1);
  head2_kernel<<<cdiv(BT_, THREADS), THREADS, 0, stream>>>(ht_f, W_t2, b_t2, out, BT_, T_, 0);
  head2_kernel<<<cdiv(BC_, THREADS), THREADS, 0, stream>>>(hc_f, W_c2, b_c2, out, BC_, C_, T_);
}
